// Network_80367428043388
// MI455X (gfx1250) — matrix operand order repaired and checked by reading, not compiled, not run on hardware
//
#include <hip/hip_runtime.h>
#include <hip/hip_bf16.h>
#include <cmath>

// ---------------------------------------------------------------------------
// MI455X / gfx1250 implementation.
//  - All large GEMMs use v_wmma_f32_16x16x32_f16 (f16 in, f32 accumulate).
//  - LSTM recurrence: ONE persistent kernel per layer (128 co-resident
//    1-wave blocks, device-scope barrier per timestep). Whh tile staged to
//    LDS once via global_load_async_to_lds_b128 (ASYNCcnt path); cell state
//    lives in registers for all 512 steps.
//  - TPR memory scan: one workgroup per batch element, 128KB LDS-resident mem.
//  - Slot attention: one wave per (s,b); lane == slot dim (wave32 == SLOT).
// ---------------------------------------------------------------------------

typedef __attribute__((ext_vector_type(16))) _Float16 v16h;
typedef __attribute__((ext_vector_type(8)))  _Float16 v8h;
typedef __attribute__((ext_vector_type(8)))  float    v8f;

#define SEQ 512
#define BAT 64
#define POS (SEQ*BAT)          // 32768
#define HID 512
#define GATES 2048
#define EMBD 256
#define RBLOCKS 128            // 4 m-tiles x 32 n-tiles for the recurrence

__device__ __forceinline__ float sigmf(float x){ return 1.f/(1.f+__expf(-x)); }
__device__ __forceinline__ float elup1(float x){ return x>0.f ? x+1.f : __expf(x); }
__device__ __forceinline__ float wred(float x){
  #pragma unroll
  for(int o=16;o;o>>=1) x += __shfl_xor(x,o,32);
  return x;
}

// ---- CDNA5 async global->LDS copy (16B per lane-op, tracked by ASYNCcnt) --
__device__ __forceinline__ void async_copy_b128(unsigned lds_byte_off,
                                                const void* gaddr){
  asm volatile("global_load_async_to_lds_b128 %0, %1, off"
               :: "v"(lds_byte_off), "v"(gaddr) : "memory");
}
__device__ __forceinline__ void wait_asynccnt0(){
  asm volatile("s_wait_asynccnt 0x0" ::: "memory");
}

// ---- WMMA fragment loaders (contiguous k = 16*half + s in both A and B) -----------------
// A (16x32 f16): lane<16 -> M=lane, K={k0..k0+15}
//                lane>=16 -> M=lane-16, K shifted by +16.
__device__ __forceinline__ v16h load_a_frag(const _Float16* __restrict__ A,
                                            int lda, int mrow, int k0, int khalf){
  const _Float16* p = A + (size_t)mrow*lda + k0 + khalf*16;
  v8h lo = *(const v8h*)(p);
  v8h hi = *(const v8h*)(p+8);
  v16h a;
  #pragma unroll
  for(int i=0;i<8;i++){ a[i]=lo[i]; a[i+8]=hi[i]; }
  return a;
}
// B (32x16 f16) from weight W[N,K] row-major: lane holds column n = n&15,
// 16 contiguous K values starting at k0 + 16*(lane>=16).
__device__ __forceinline__ v16h load_b_frag(const _Float16* __restrict__ W,
                                            int ldw, int nrow, int k0, int khalf){
  return *(const v16h*)(W + (size_t)nrow*ldw + k0 + khalf*16);
}

// ---- Generic GEMM: C[M,N] = A[M,K](f16) @ W[N,K]^T(f16) + b1 + b2 ---------
__global__ __launch_bounds__(32)
void gemm_f16_wmma(const _Float16* __restrict__ A, const _Float16* __restrict__ W,
                   const float* __restrict__ b1, const float* __restrict__ b2,
                   float* __restrict__ C, int N, int K, int lda){
  const int lane  = threadIdx.x;
  const int m0    = blockIdx.x * 16;
  const int n0    = blockIdx.y * 64;
  const int mrow  = m0 + (lane & 15);
  const int khalf = lane >> 4;
  const int ncol  = lane & 15;
  v8f acc[4] = {};
  for(int k0 = 0; k0 < K; k0 += 32){
    v16h a = load_a_frag(A, lda, mrow, k0, khalf);
    #pragma unroll
    for(int j=0;j<4;j++){
      v16h b = load_b_frag(W, K, n0 + j*16 + ncol, k0, khalf);
      acc[j] = __builtin_amdgcn_wmma_f32_16x16x32_f16(
          false, a, false, b, (short)0, acc[j], false, false);
    }
  }
  const int rbase = khalf * 8;
  #pragma unroll
  for(int j=0;j<4;j++){
    const int cc = n0 + j*16 + ncol;
    float bias = (b1 ? b1[cc] : 0.f) + (b2 ? b2[cc] : 0.f);
    #pragma unroll
    for(int r=0;r<8;r++){
      const int row = m0 + rbase + r;
      C[(size_t)row*N + cc] = acc[j][r] + bias;
    }
  }
}

// ---- device-scope barrier for the persistent recurrence -------------------
__device__ __forceinline__ void grid_sync(unsigned* cnt, unsigned target){
  __threadfence();
  if(threadIdx.x == 0){
    __hip_atomic_fetch_add(cnt, 1u, __ATOMIC_ACQ_REL, __HIP_MEMORY_SCOPE_AGENT);
    while(__hip_atomic_load(cnt, __ATOMIC_ACQUIRE, __HIP_MEMORY_SCOPE_AGENT) < target){
      __builtin_amdgcn_s_sleep(2);
    }
  }
  __syncthreads();
}

// ---- Persistent LSTM layer: all 512 timesteps in one launch ---------------
// grid = 128 blocks x 32 threads. Block (mb = bid&3, nb = bid>>2) owns the
// h-tile [mb*16..+16) x [nb*16..+16) and its 4 gate tiles. Whh working set
// (64 rows x 512 K = 64KB f16) is async-staged into LDS once.
__global__ __launch_bounds__(32)
void lstm_layer_kernel(const float* __restrict__ X,      // [SEQ,64,2048] biases folded
                       const _Float16* __restrict__ Whh, // [2048,512]
                       _Float16* __restrict__ Hbase,     // [(SEQ+1), 64, ldh]
                       int ldh,
                       unsigned* __restrict__ bar){
  const int lane  = threadIdx.x;
  const int mb    = blockIdx.x & 3;
  const int nb    = blockIdx.x >> 2;
  const int m0    = mb * 16;
  const int mrow  = m0 + (lane & 15);
  const int khalf = lane >> 4;
  const int ncol  = lane & 15;
  const int col   = nb*16 + ncol;
  const int rbase = khalf * 8;

  extern __shared__ float dynsm[];
  _Float16* wsh = (_Float16*)dynsm;            // [64 rows][512 K] = 64KB
  const unsigned lds_base = (unsigned)(unsigned long long)(const void*)wsh;

  // ---- one-time async stage of Whh tile: 4096 x 16B chunks ----
  for(int chunk = lane; chunk < 4096; chunk += 32){
    const int lrow = chunk >> 6;               // 64 x 16B chunks per row
    const int coff = (chunk & 63) * 16;        // byte offset within row
    const int g = lrow >> 4, r = lrow & 15;
    const char* src = (const char*)(Whh + (size_t)(g*HID + nb*16 + r)*HID) + coff;
    async_copy_b128(lds_base + (unsigned)(lrow*1024 + coff), src);
  }
  wait_asynccnt0();
  __syncthreads();

  float cstate[8];
  #pragma unroll
  for(int r=0;r<8;r++) cstate[r]=0.f;

  for(int s=0; s<SEQ; s++){
    const float*    Xs = X     + (size_t)s*BAT*GATES;
    const _Float16* Hp = Hbase + (size_t)s*BAT*ldh;
    v8f acc[4] = {};
    for(int k0=0; k0<HID; k0+=32){
      v16h a = load_a_frag(Hp, ldh, mrow, k0, khalf);
      #pragma unroll
      for(int g=0; g<4; g++){
        v16h b = *(const v16h*)(wsh + (size_t)(g*16 + ncol)*HID + k0 + khalf*16);
        acc[g] = __builtin_amdgcn_wmma_f32_16x16x32_f16(
            false, a, false, b, (short)0, acc[g], false, false);
      }
    }
    _Float16* Ho = Hbase + (size_t)(s+1)*BAT*ldh;
    #pragma unroll
    for(int r=0;r<8;r++){
      const int row = m0 + rbase + r;
      const float* xr = X ? (Xs + (size_t)row*GATES) : nullptr;
      float zi = acc[0][r] + xr[col];
      float zf = acc[1][r] + xr[HID   + col];
      float zg = acc[2][r] + xr[2*HID + col];
      float zo = acc[3][r] + xr[3*HID + col];
      float cn = sigmf(zf)*cstate[r] + sigmf(zi)*tanhf(zg);
      float h  = sigmf(zo)*tanhf(cn);
      cstate[r] = cn;
      Ho[(size_t)row*ldh + col] = (_Float16)h;
    }
    grid_sync(bar, (unsigned)RBLOCKS*(unsigned)(s+1));
  }
}

// ---- Embedding gather -> f16 ---------------------------------------------
__global__ void embed_gather_kernel(const int* __restrict__ tok,
                                    const float* __restrict__ embW,
                                    _Float16* __restrict__ out){
  int i = blockIdx.x*blockDim.x + threadIdx.x;   // over POS*EMBD
  int row = i >> 8, k = i & 255;
  int t = tok[row];
  out[i] = (_Float16)embW[(size_t)t*EMBD + k];
}

// ---- f32 -> f16 convert ---------------------------------------------------
__global__ void cvt_f32_f16(const float* __restrict__ s, _Float16* __restrict__ d, int n){
  int i = blockIdx.x*blockDim.x + threadIdx.x;
  if(i < n) d[i] = (_Float16)s[i];
}

// ---- Slot attention: one wave per (s,b) position --------------------------
__global__ __launch_bounds__(32)
void slot_attn_kernel(const float* __restrict__ dec,   // [POS,192]
                      const float* __restrict__ Wi, const float* __restrict__ bi, int nsl,
                      const float* __restrict__ Wq, const float* __restrict__ bq,
                      const float* __restrict__ Wk, const float* __restrict__ bk,
                      const float* __restrict__ Wv, const float* __restrict__ bv,
                      const float* __restrict__ lng, const float* __restrict__ lnb,
                      const float* __restrict__ Wm1, const float* __restrict__ bm1,
                      const float* __restrict__ Wm2, const float* __restrict__ bm2,
                      const float* __restrict__ Wsp, const float* __restrict__ bsp,
                      float* __restrict__ out){
  const int p = blockIdx.x;
  const int d = threadIdx.x;
  __shared__ float inp[192];
  __shared__ float slds[96];
  __shared__ float ilds[96];
  __shared__ float hl[32];
  __shared__ float hidl[64];
  __shared__ float lg[9];
  const float* row = dec + (size_t)p*192;
  for(int k=d;k<192;k+=32) inp[k]=row[k];
  __syncthreads();

  float init[3], slot[3], km[3], vm[3];
  for(int j=0;j<nsl;j++){
    float acc = bi[j*32+d];
    const float* w = Wi + (size_t)(j*32+d)*192;
    for(int k=0;k<192;k++) acc += w[k]*inp[k];
    init[j]=acc; slot[j]=acc; ilds[j*32+d]=acc;
  }
  for(int i=0;i<3;i++){
    float ka=bk[d], va=bv[d];
    for(int k=0;k<64;k++){ float x=inp[i*64+k]; ka+=Wk[d*64+k]*x; va+=Wv[d*64+k]*x; }
    km[i]=elup1(ka); vm[i]=va;
  }
  const float rs = 0.17677669529663687f;  // 32^-0.5
  for(int it=0; it<3; ++it){
    __syncthreads();
    for(int j=0;j<nsl;j++) slds[j*32+d]=slot[j];
    __syncthreads();
    float q[3];
    for(int j=0;j<nsl;j++){
      float acc = bq[d] + init[j];
      for(int k=0;k<32;k++) acc += Wq[d*32+k]*slds[j*32+k];
      q[j]=elup1(acc*rs);
    }
    for(int i=0;i<3;i++)
      for(int j=0;j<nsl;j++){
        float v = wred(km[i]*q[j]);
        if(d==0) lg[i*3+j]=v;
      }
    __syncthreads();
    float attn[3][3];
    for(int i=0;i<3;i++){
      float mx=-1e30f;
      for(int j=0;j<nsl;j++) mx=fmaxf(mx,lg[i*3+j]);
      float s=0.f;
      for(int j=0;j<nsl;j++){ attn[i][j]=__expf(lg[i*3+j]-mx); s+=attn[i][j]; }
      for(int j=0;j<nsl;j++) attn[i][j]=attn[i][j]/s + 1e-8f;
    }
    for(int j=0;j<nsl;j++){
      float cs=0.f; for(int i=0;i<3;i++) cs+=attn[i][j];
      for(int i=0;i<3;i++) attn[i][j]/=cs;
    }
    for(int j=0;j<nsl;j++){
      float upd=0.f;
      for(int i=0;i<3;i++) upd += attn[i][j]*vm[i];
      float mn  = wred(upd)*(1.f/32.f);
      float ctr = upd-mn;
      float vv  = wred(ctr*ctr)*(1.f/32.f);
      float h   = ctr*rsqrtf(vv+1e-5f)*lng[d]+lnb[d];
      __syncthreads();
      hl[d]=h;
      __syncthreads();
      for(int m=d;m<64;m+=32){
        float a=bm1[m];
        for(int k=0;k<32;k++) a += Wm1[m*32+k]*hl[k];
        hidl[m]=fmaxf(a,0.f);
      }
      __syncthreads();
      float o=bm2[d];
      for(int m=0;m<64;m++) o += Wm2[d*64+m]*hidl[m];
      slot[j] += o*(1.f/32.f);
      __syncthreads();
    }
  }
  __syncthreads();
  for(int j=0;j<nsl;j++) slds[j*32+d]=slot[j];
  __syncthreads();
  for(int j=0;j<nsl;j++){
    float acc=bsp[d];
    for(int k=0;k<32;k++)
      acc += Wsp[d*64+k]*ilds[j*32+k] + Wsp[d*64+32+k]*slds[j*32+k];
    out[(size_t)p*nsl*32 + j*32 + d]=acc;
  }
}

// ---- mask reads (role1/role2/filler/u1/u2) --------------------------------
__global__ void maskread_kernel(const float* __restrict__ bind,
                                const float* __restrict__ reas,
                                float* r1,float* r2,float* fl,float* u1,float* u2){
  int i = blockIdx.x*blockDim.x + threadIdx.x;    // POS*32
  int p = i>>5, d = i&31;
  const float e=1e-6f, a=1.f-2e-6f;
  float b0=bind[(size_t)p*96+d], b1v=bind[(size_t)p*96+32+d], b2v=bind[(size_t)p*96+64+d];
  r1[i]=tanhf(a*b0+e*b1v+e*b2v);
  r2[i]=tanhf(e*b0+a*b1v+e*b2v);
  fl[i]=tanhf(e*b0+e*b1v+a*b2v);
  float s0=reas[(size_t)p*64+d], s1=reas[(size_t)p*64+32+d];
  u1[i]=tanhf(a*s0+e*s1);
  u2[i]=tanhf(e*s0+a*s1);
}

// ---- gate = sigmoid(h1 . Wg + bg + 1) -------------------------------------
__global__ __launch_bounds__(256)
void gate_kernel(const _Float16* __restrict__ cat, const float* __restrict__ Wg,
                 const float* __restrict__ bg, float* __restrict__ gate){
  int wid = threadIdx.x>>5, lane = threadIdx.x&31;
  int pos = blockIdx.x*8 + wid;
  const _Float16* h = cat + (size_t)(BAT+pos)*544;
  float a=0.f;
  for(int k=lane;k<HID;k+=32) a += (float)h[k]*Wg[k];
  a = wred(a);
  if(lane==0) gate[pos] = sigmf(a + bg[0] + 1.f);
}

// ---- TPR memory scan: LDS-resident 32^3 mem, one block per batch elem -----
__global__ __launch_bounds__(256)
void scan_kernel(const float* __restrict__ role1, const float* __restrict__ role2,
                 const float* __restrict__ filler,
                 const float* __restrict__ u1, const float* __restrict__ u2,
                 const float* __restrict__ gate, _Float16* __restrict__ cat){
  extern __shared__ float dynsm[];
  float* mem  = dynsm;              // 32768 floats = 128KB
  float* red  = dynsm + 32768;      // 256
  float* lr1  = red + 256;          // 32
  float* lr2  = lr1 + 32;
  float* lf   = lr2 + 32;
  float* lu1  = lf  + 32;
  float* lu2  = lu1 + 32;
  float* curv = lu2 + 32;
  float* sc   = curv + 32;          // [0]=scale, [1]=gate
  const int b   = blockIdx.x;
  const int tid = threadIdx.x;
  const int v   = tid & 31;
  const int grp = tid >> 5;
  for(int kk=0;kk<128;kk++) mem[(grp*128+kk)*32+v]=0.f;
  __syncthreads();
  for(int s=0;s<SEQ;s++){
    size_t pb = (size_t)(s*BAT+b)*32;
    if(tid<32){
      lr1[tid]=role1[pb+tid]; lr2[tid]=role2[pb+tid]; lf[tid]=filler[pb+tid];
      lu1[tid]=u1[pb+tid];    lu2[tid]=u2[pb+tid];
    }
    if(tid==32) sc[1]=gate[s*BAT+b];
    __syncthreads();
    float pp=0.f;
    for(int kk=0;kk<128;kk++){
      int rt=grp*128+kk;
      pp += lr1[rt>>5]*lr2[rt&31]*mem[rt*32+v];
    }
    red[grp*32+v]=pp;
    __syncthreads();
    if(grp==0){
      float prev=0.f;
      for(int g8=0;g8<8;g8++) prev+=red[g8*32+v];
      curv[v] = sc[1]*(lf[v]-prev)*(1.f/32.f);   // cur/MEM folded in
    }
    __syncthreads();
    float ss=0.f;
    for(int kk=0;kk<128;kk++){
      int rt=grp*128+kk;
      float m2 = mem[rt*32+v] + lr1[rt>>5]*lr2[rt&31]*curv[v];
      mem[rt*32+v]=m2; ss+=m2*m2;
    }
    ss = wred(ss);
    if(v==0) red[grp]=ss;
    __syncthreads();
    if(tid==0){
      float t=0.f; for(int g8=0;g8<8;g8++) t+=red[g8];
      float nr = fmaxf(sqrtf(t)-1.f,0.f)+1.f;
      sc[0]=1.f/nr;
    }
    __syncthreads();
    float scale=sc[0], rp=0.f;
    for(int kk=0;kk<128;kk++){
      int rt=grp*128+kk;
      float m2 = mem[rt*32+v]*scale;
      mem[rt*32+v]=m2;
      rp += lu1[rt>>5]*lu2[rt&31]*m2;
    }
    red[grp*32+v]=rp;
    __syncthreads();
    if(grp==0){
      float rv=0.f;
      for(int g8=0;g8<8;g8++) rv+=red[g8*32+v];
      float mn = wred(rv)*(1.f/32.f);
      float ct = rv-mn;
      float vv = wred(ct*ct)*(1.f/32.f);
      cat[(size_t)(BAT + s*BAT + b)*544 + HID + v] = (_Float16)(ct*rsqrtf(vv+1e-5f));
    }
    __syncthreads();
  }
}

// ---------------------------------------------------------------------------
extern "C" void kernel_launch(void* const* d_in, const int* in_sizes, int n_in,
                              void* d_out, int out_size, void* d_ws, size_t ws_size,
                              hipStream_t stream){
  (void)in_sizes; (void)n_in; (void)out_size; (void)ws_size;
  const int*   tokens = (const int*)  d_in[0];
  const float* embW   = (const float*)d_in[1];
  const float* Wih0   = (const float*)d_in[2];
  const float* Whh0   = (const float*)d_in[3];
  const float* bih0   = (const float*)d_in[4];
  const float* bhh0   = (const float*)d_in[5];
  const float* Wih1   = (const float*)d_in[6];
  const float* Whh1   = (const float*)d_in[7];
  const float* bih1   = (const float*)d_in[8];
  const float* bhh1   = (const float*)d_in[9];
  const float* Wpi    = (const float*)d_in[10];
  const float* bpi    = (const float*)d_in[11];
  const float* Wq     = (const float*)d_in[12];
  const float* bq     = (const float*)d_in[13];
  const float* Wk     = (const float*)d_in[14];
  const float* bk     = (const float*)d_in[15];
  const float* Wv     = (const float*)d_in[16];
  const float* bv     = (const float*)d_in[17];
  const float* lng    = (const float*)d_in[18];
  const float* lnb    = (const float*)d_in[19];
  const float* Wm1    = (const float*)d_in[20];
  const float* bm1    = (const float*)d_in[21];
  const float* Wm2    = (const float*)d_in[22];
  const float* bm2    = (const float*)d_in[23];
  const float* Wbind  = (const float*)d_in[24];
  const float* bbind  = (const float*)d_in[25];
  const float* Wreas  = (const float*)d_in[26];
  const float* breas  = (const float*)d_in[27];
  const float* Wsp    = (const float*)d_in[28];
  const float* bsp    = (const float*)d_in[29];
  const float* Wg     = (const float*)d_in[30];
  const float* bg     = (const float*)d_in[31];
  const float* Wout   = (const float*)d_in[32];
  const float* bout   = (const float*)d_in[33];
  float* outp = (float*)d_out;

  // ---- workspace bump allocator ----
  char* ws = (char*)d_ws; size_t off = 0;
  auto alloc = [&](size_t bytes)->void*{
    void* p = ws + off;
    off = (off + bytes + 255) & ~(size_t)255;
    return p;
  };
  _Float16* embh  = (_Float16*)alloc((size_t)POS*EMBD*2);
  _Float16* Wih0h = (_Float16*)alloc((size_t)GATES*EMBD*2);
  _Float16* Whh0h = (_Float16*)alloc((size_t)GATES*HID*2);
  _Float16* Wih1h = (_Float16*)alloc((size_t)GATES*HID*2);
  _Float16* Whh1h = (_Float16*)alloc((size_t)GATES*HID*2);
  _Float16* Wpih  = (_Float16*)alloc((size_t)192*EMBD*2);
  _Float16* Wouth = (_Float16*)alloc((size_t)128*544*2);
  float*    X     = (float*)   alloc((size_t)POS*GATES*4);      // reused layer0/layer1
  _Float16* hbf0  = (_Float16*)alloc((size_t)(SEQ+1)*BAT*HID*2);
  _Float16* cat   = (_Float16*)alloc((size_t)(SEQ+1)*BAT*544*2);
  float*    dec   = (float*)   alloc((size_t)POS*192*4);
  float*    bind  = (float*)   alloc((size_t)POS*96*4);
  float*    reas  = (float*)   alloc((size_t)POS*64*4);
  float*    role1 = (float*)   alloc((size_t)POS*32*4);
  float*    role2 = (float*)   alloc((size_t)POS*32*4);
  float*    fill  = (float*)   alloc((size_t)POS*32*4);
  float*    u1    = (float*)   alloc((size_t)POS*32*4);
  float*    u2    = (float*)   alloc((size_t)POS*32*4);
  float*    gate  = (float*)   alloc((size_t)POS*4);
  unsigned* bar0  = (unsigned*)alloc(256);
  unsigned* bar1  = (unsigned*)alloc(256);

  // ---- weight conversions + embedding gather ----
  auto cvt = [&](const float* s, _Float16* d, int n){
    cvt_f32_f16<<<(n+255)/256, 256, 0, stream>>>(s, d, n);
  };
  cvt(Wih0, Wih0h, GATES*EMBD);
  cvt(Whh0, Whh0h, GATES*HID);
  cvt(Wih1, Wih1h, GATES*HID);
  cvt(Whh1, Whh1h, GATES*HID);
  cvt(Wpi,  Wpih,  192*EMBD);
  cvt(Wout, Wouth, 128*544);
  embed_gather_kernel<<<(POS*EMBD)/256, 256, 0, stream>>>(tokens, embW, embh);

  // zero h0 state, first cat slice (h at t=-1) and the two step-barriers
  hipMemsetAsync(hbf0, 0, (size_t)BAT*HID*2, stream);
  hipMemsetAsync(cat,  0, (size_t)BAT*544*2, stream);
  hipMemsetAsync(bar0, 0, 256, stream);
  hipMemsetAsync(bar1, 0, 256, stream);

  const int lstm_lds = 64*HID*2;   // 64KB Whh tile per block
  hipFuncSetAttribute(reinterpret_cast<const void*>(lstm_layer_kernel),
                      hipFuncAttributeMaxDynamicSharedMemorySize, lstm_lds);

  // ---- layer 0: X = emb @ Wih0^T + bih0 + bhh0, persistent recurrence -----
  gemm_f16_wmma<<<dim3(POS/16, GATES/64), 32, 0, stream>>>(
      embh, Wih0h, bih0, bhh0, X, GATES, EMBD, EMBD);
  lstm_layer_kernel<<<RBLOCKS, 32, lstm_lds, stream>>>(
      X, Whh0h, hbf0, HID, bar0);

  // ---- layer 1: X = h0 @ Wih1^T + bih1 + bhh1, h1 written f16 into cat ----
  gemm_f16_wmma<<<dim3(POS/16, GATES/64), 32, 0, stream>>>(
      hbf0 + (size_t)BAT*HID, Wih1h, bih1, bhh1, X, GATES, HID, HID);
  lstm_layer_kernel<<<RBLOCKS, 32, lstm_lds, stream>>>(
      X, Whh1h, cat, 544, bar1);

  // ---- decoder input projection + slot attention (bind / reason) ----------
  gemm_f16_wmma<<<dim3(POS/16, 192/64), 32, 0, stream>>>(
      embh, Wpih, bpi, nullptr, dec, 192, EMBD, EMBD);
  slot_attn_kernel<<<POS, 32, 0, stream>>>(
      dec, Wbind, bbind, 3, Wq,bq, Wk,bk, Wv,bv, lng,lnb, Wm1,bm1, Wm2,bm2,
      Wsp,bsp, bind);
  slot_attn_kernel<<<POS, 32, 0, stream>>>(
      dec, Wreas, breas, 2, Wq,bq, Wk,bk, Wv,bv, lng,lnb, Wm1,bm1, Wm2,bm2,
      Wsp,bsp, reas);
  maskread_kernel<<<(POS*32)/256, 256, 0, stream>>>(bind, reas, role1, role2, fill, u1, u2);
  gate_kernel<<<POS/8, 256, 0, stream>>>(cat, Wg, bg, gate);

  // ---- TPR memory scan: 64 blocks, 128KB+ dynamic LDS each ----------------
  const int scan_lds = (32768 + 256 + 32*7 + 8) * 4;   // ~133 KB
  hipFuncSetAttribute(reinterpret_cast<const void*>(scan_kernel),
                      hipFuncAttributeMaxDynamicSharedMemorySize, scan_lds);
  scan_kernel<<<BAT, 256, scan_lds, stream>>>(role1, role2, fill, u1, u2, gate, cat);

  // ---- output projection: [POS,544] @ Wout^T -> [POS,128] -----------------
  gemm_f16_wmma<<<dim3(POS/16, 128/64), 32, 0, stream>>>(
      cat + (size_t)BAT*544, Wouth, bout, nullptr, outp, 128, 544, 544);
}
